// GCLSTMPrisoner_50766513439412
// MI455X (gfx1250) — matrix operand order repaired and checked by reading, not compiled, not run on hardware
//
#include <hip/hip_runtime.h>
#include <hip/hip_bf16.h>

typedef _Float16 half_t;
typedef __attribute__((ext_vector_type(16))) _Float16 v16h;
typedef __attribute__((ext_vector_type(8)))  _Float16 v8h;
typedef __attribute__((ext_vector_type(8)))  float    v8f;

#define T_STEPS 128
#define AGENTS  16
#define F_IN    32
#define HID     384
#define G4H     (4 * HID)        // 1536
#define KSTEPS  13               // 416 / 32
#define CTILES  96               // 1536 / 16
#define ROWS    16               // node rows per workgroup (= one batch element)
#define NWG     64               // 1024 / 16
#define NTHREADS 384             // 12 waves
#define ASTRIDE 424              // padded LDS row stride (f16 elems), 16B aligned
#define OUTC    387              // 384 + 2 + 1

// v16h units for B-fragment addressing:
//   addr(halfs) = g*159744 + ut*6656 + kk*512 + lane*16
#define GSTRIDE_V 9984           // 159744 / 16
#define KSTRIDE_V 32             // 512 / 16

// ---------------------------------------------------------------------------
// Setup: pack f32 [Wx ; Wh] (416 x 1536) into f16 WMMA B-fragments.
// Fragment (ct, kk) = 32x16 (KxN) tile in per-lane register order of
// v_wmma_f32_16x16x32_f16's B operand:
//   lane l, slot j -> K = kk*32 + (l>=16 ? 8 : 0) + (j<8 ? j : j+8),  N = ct*16 + (l&15)
// Memory: frag = ct*13 + kk, addr = (frag*32 + lane)*16 halfs.
// ---------------------------------------------------------------------------
__global__ __launch_bounds__(256)
void build_wfrag_kernel(const float* __restrict__ Wx,
                        const float* __restrict__ Wh,
                        half_t* __restrict__ Wfrag) {
  int gid = blockIdx.x * blockDim.x + threadIdx.x;
  const int total = KSTEPS * CTILES * 32;
  if (gid >= total) return;
  int lane = gid & 31;
  int frag = gid >> 5;               // ct*13 + kk
  int kk   = frag % KSTEPS;
  int ct   = frag / KSTEPS;
  int col  = ct * 16 + (lane & 15);
  int kbase = kk * 32 + ((lane >> 4) << 3);
  v16h v;
#pragma unroll
  for (int j = 0; j < 16; ++j) {
    int K = kbase + (j < 8 ? j : j + 8);
    float w = (K < F_IN) ? Wx[(size_t)K * G4H + col]
                         : Wh[(size_t)(K - F_IN) * G4H + col];
    v[j] = (half_t)w;
  }
  *(v16h*)(Wfrag + (size_t)gid * 16) = v;
}

__device__ __forceinline__ float sigmoid_f(float x) {
  return 1.0f / (1.0f + __expf(-x));
}

__device__ __forceinline__ void load_bfrags(const v16h* __restrict__ wbase,
                                            int kk, v16h b[4]) {
#pragma unroll
  for (int g = 0; g < 4; ++g) b[g] = wbase[g * GSTRIDE_V + kk * KSTRIDE_V];
}

__device__ __forceinline__ v16h load_afrag(const half_t* Ab, int off) {
  v8h alo = *(const v8h*)(Ab + off);
  v8h ahi = *(const v8h*)(Ab + off + 16);
  return __builtin_shufflevector(alo, ahi, 0, 1, 2, 3, 4, 5, 6, 7,
                                 8, 9, 10, 11, 12, 13, 14, 15);
}

// ---------------------------------------------------------------------------
// Main: 64 WGs, each owns 16 node rows = one batch element, full 128-step
// recurrence per WG, zero cross-WG traffic. LDS double-buffers the f16
// A = [x_t | h] operand; cell state c lives in registers (lane-owned per the
// WMMA C/D layout). 12 waves; wave w owns hidden tiles ut = w and w+12:
// 24 tiles x 4 gates x 13 K-steps = 1248 v_wmma_f32_16x16x32_f16 per WG/step.
// Weight fragments are re-streamed from the 192MB L2 every step (1.22 MB/WG).
// An opaque per-step integer offset blocks LICM from hoisting (and spilling)
// the weight loads, while keeping the base a known-global pointer so they
// lower to global_load_b128 (LOADcnt only — decoupled from the LDS DScnt
// stream) with immediate offsets.
// ---------------------------------------------------------------------------
__global__ __launch_bounds__(NTHREADS)
void gclstm_main_kernel(const float* __restrict__ agent_obs,
                        const float* __restrict__ hideout_obs,
                        const float* __restrict__ timestep_obs,
                        const float* __restrict__ b_in,
                        const float* __restrict__ b_conv,
                        const float* __restrict__ w_ci,
                        const float* __restrict__ w_cf,
                        const float* __restrict__ w_co,
                        const half_t* __restrict__ Wfrag,
                        float* __restrict__ out) {
  __shared__ __align__(16) half_t Abuf[2][ROWS * ASTRIDE];
  __shared__ float biasS[4 * HID];
  __shared__ float peepS[3 * HID];

  const int tid   = threadIdx.x;
  const int wg    = blockIdx.x;        // batch element
  const int lane  = tid & 31;
  const int wave  = tid >> 5;          // 0..11
  const int nhalf = lane >> 4;         // 0/1
  const int ncol  = lane & 15;         // N within tile

  // bias = b + b_conv ; peephole weights
  for (int j = tid; j < 4 * HID; j += NTHREADS) biasS[j] = b_in[j] + b_conv[j];
  for (int j = tid; j < HID; j += NTHREADS) {
    peepS[j]           = w_ci[j];
    peepS[HID + j]     = w_cf[j];
    peepS[2 * HID + j] = w_co[j];
  }
  // h0 = 0 in buffer 0's h-region
  for (int j = tid; j < ROWS * HID; j += NTHREADS) {
    int r = j / HID, u = j - r * HID;
    Abuf[0][r * ASTRIDE + F_IN + u] = (half_t)0.0f;
  }

  float creg[2][8];   // cell state: pair p, M = nhalf*8 + r, N = ncol
#pragma unroll
  for (int p = 0; p < 2; ++p)
#pragma unroll
    for (int r = 0; r < 8; ++r) creg[p][r] = 0.0f;

  const int arow = ncol * ASTRIDE;      // this lane's A row (M = lane&15)
  const int kofs = nhalf * 8;           // A-layout K offset per lane half
  // x_t for this WG is a contiguous 512-float block per step
  const float* xsrc = agent_obs + (size_t)wg * T_STEPS * ROWS * F_IN;
  int wofs = 0;                         // always 0, but opaque per iteration

  __syncthreads();

  for (int t = 0; t < T_STEPS; ++t) {
    // Anti-LICM: "redefine" the (always-zero) weight offset every iteration.
    // The base pointer stays the kernarg, so address-space inference keeps
    // the loads on the GLOBAL path with immediate offsets.
    asm volatile("" : "+s"(wofs));
    const half_t* Wt = Wfrag + wofs;

    const int buf = t & 1;
    // stage x_t (contiguous 16x32 f32 block -> f16); waves 0..7, 2 elts each
    if (tid < 256) {
      const float* xt = xsrc + (size_t)t * (ROWS * F_IN);
#pragma unroll
      for (int q = 0; q < 2; ++q) {
        int idx = tid + q * 256;
        Abuf[buf][(idx >> 5) * ASTRIDE + (idx & 31)] = (half_t)xt[idx];
        __builtin_prefetch(xt + ROWS * F_IN + idx, 0, 0);  // next step's x
      }
    }
    __syncthreads();  // x_t staged AND previous step's h writes visible

    const half_t* Ab = &Abuf[buf][0];
    half_t*       Aw = &Abuf[buf ^ 1][0];

#pragma unroll
    for (int p = 0; p < 2; ++p) {
      const int ut = wave + 12 * p;     // 0..23
      // single base pointer; every fragment is an immediate offset from it
      const v16h* wbase =
          (const v16h*)(Wt + (size_t)ut * 6656 + (size_t)lane * 16);
      v8f acc[4] = {v8f{}, v8f{}, v8f{}, v8f{}};

      // software pipeline: B fragments 2 k-groups ahead, A 1 ahead
      v16h bq[3][4];
      v16h aq[2];
      load_bfrags(wbase, 0, bq[0]);
      load_bfrags(wbase, 1, bq[1]);
      aq[0] = load_afrag(Ab, arow + kofs);
#pragma unroll
      for (int kk = 0; kk < KSTEPS; ++kk) {
        if (kk + 2 < KSTEPS) load_bfrags(wbase, kk + 2, bq[(kk + 2) % 3]);
        if (kk + 1 < KSTEPS)
          aq[(kk + 1) & 1] = load_afrag(Ab, arow + (kk + 1) * 32 + kofs);
#pragma unroll
        for (int g = 0; g < 4; ++g) {
          acc[g] = __builtin_amdgcn_wmma_f32_16x16x32_f16(
              false, aq[kk & 1], false, bq[kk % 3][g], (short)0, acc[g],
              false, false);
        }
      }

      // LSTM elementwise update (bias/peephole from LDS: keeps them out of
      // the GEMM loop's register budget)
      const int u   = ut * 16 + ncol;
      const float bi  = biasS[u];
      const float bff = biasS[HID + u];
      const float bc  = biasS[2 * HID + u];
      const float bo  = biasS[3 * HID + u];
      const float pci = peepS[u];
      const float pcf = peepS[HID + u];
      const float pco = peepS[2 * HID + u];
      half_t* hdst = Aw + (nhalf * 8) * ASTRIDE + F_IN + u;
#pragma unroll
      for (int r = 0; r < 8; ++r) {
        float c  = creg[p][r];
        float ii = sigmoid_f(acc[0][r] + bi + pci * c);
        float ff = sigmoid_f(acc[1][r] + bff + pcf * c);
        float cn = ff * c + ii * tanhf(acc[2][r] + bc);
        float oo = sigmoid_f(acc[3][r] + bo + pco * cn);
        float hn = oo * tanhf(cn);
        creg[p][r] = cn;
        hdst[r * ASTRIDE] = (half_t)hn;
      }
    }
  }
  __syncthreads();

  // global_mean_pool over the 16 agents: final h is in Abuf[0] (t=127 wrote
  // buf^1 = 0). Pool the same f16 values the recurrence consumed.
  for (int u = tid; u < HID; u += NTHREADS) {
    float s = 0.0f;
#pragma unroll
    for (int r = 0; r < ROWS; ++r)
      s += (float)Abuf[0][r * ASTRIDE + F_IN + u];
    out[(size_t)wg * OUTC + u] = s * (1.0f / 16.0f);
  }
  // concat hideout_obs (2) and timestep_obs (1)
  if (tid < 3) {
    out[(size_t)wg * OUTC + 384 + tid] =
        (tid < 2) ? hideout_obs[wg * 2 + tid] : timestep_obs[wg];
  }
}

// ---------------------------------------------------------------------------
extern "C" void kernel_launch(void* const* d_in, const int* in_sizes, int n_in,
                              void* d_out, int out_size, void* d_ws, size_t ws_size,
                              hipStream_t stream) {
  const float* agent_obs    = (const float*)d_in[0];
  const float* hideout_obs  = (const float*)d_in[1];
  const float* timestep_obs = (const float*)d_in[2];
  const float* Wx           = (const float*)d_in[3];
  const float* Wh           = (const float*)d_in[4];
  const float* b_in         = (const float*)d_in[5];
  const float* b_conv       = (const float*)d_in[6];
  const float* w_ci         = (const float*)d_in[7];
  const float* w_cf         = (const float*)d_in[8];
  const float* w_co         = (const float*)d_in[9];
  float* out                = (float*)d_out;

  half_t* Wfrag = (half_t*)d_ws;  // 416*1536*2 B ~= 1.22 MB of scratch

  const int total = KSTEPS * CTILES * 32;            // 39936 threads
  build_wfrag_kernel<<<(total + 255) / 256, 256, 0, stream>>>(Wx, Wh, Wfrag);

  gclstm_main_kernel<<<NWG, NTHREADS, 0, stream>>>(
      agent_obs, hideout_obs, timestep_obs, b_in, b_conv,
      w_ci, w_cf, w_co, Wfrag, out);
}
